// AdaptiveSoftmax_12515534700713
// MI455X (gfx1250) — matrix operand order repaired and checked by reading, not compiled, not run on hardware
//
#include <hip/hip_runtime.h>
#include <math.h>

typedef __attribute__((ext_vector_type(16))) _Float16 v16h;
typedef __attribute__((ext_vector_type(8)))  _Float16 v8h;
typedef __attribute__((ext_vector_type(8)))  float    v8f;

#define ROWS      2048      // B*T
#define DMODEL    1024
#define HEAD_N    2002
#define HEAD_NPAD 2048      // multiple of 128
#define CUT0      2000
#define TAIL0_N   8000
#define TAIL0_NP  8064      // multiple of 128
#define TAIL1_N   40257
#define TAIL1_NP  40320     // multiple of 128
#define VOCAB     50257

// ---------------- conversion kernels ----------------

__global__ void cvt_f32_to_f16(const float* __restrict__ in, _Float16* __restrict__ out, int n) {
  int i = blockIdx.x * blockDim.x + threadIdx.x;
  if (i < n) out[i] = (_Float16)in[i];
}

// in: [K x N] f32 row-major -> out: [Npad x K] f16 row-major (transposed, zero padded)
__global__ void cvt_transpose_f16(const float* __restrict__ in, _Float16* __restrict__ out,
                                  int K, int N, int Npad) {
  int i = blockIdx.x * blockDim.x + threadIdx.x;
  if (i >= Npad * K) return;
  int n = i / K;
  int k = i - n * K;
  out[i] = (n < N) ? (_Float16)in[(size_t)k * N + n] : (_Float16)0.0f;
}

// ---------------- WMMA GEMM ----------------
// C[M x Nreal](f32, optional, strided into ldOut with column offset colOff)
//   = A[M x K](f16 row-major) * Bt[Npad x K](f16, B transposed) + bias
// Hout (optional): f16 [M x Npad] copy of the result (feeds next GEMM).
// Each wave computes a 16(M) x (16*NS)(N) tile: one A fragment reused by NS WMMAs per K-step.
template <int NS>
__global__ __launch_bounds__(256) void gemm_wmma_f16(
    const _Float16* __restrict__ A, const _Float16* __restrict__ Bt,
    const float* __restrict__ bias, float* __restrict__ Cout,
    _Float16* __restrict__ Hout,
    int M, int K, int Npad, int Nreal, int ldOut, int colOff)
{
  int tilesN = Npad / (16 * NS);
  int tilesM = M >> 4;
  int wave = blockIdx.x * (blockDim.x >> 5) + (threadIdx.x >> 5);
  if (wave >= tilesM * tilesN) return;
  int tm = wave / tilesN;
  int tn = wave - tm * tilesN;
  int mBase = tm << 4;
  int nBase = tn * (16 * NS);
  int lane = threadIdx.x & 31;
  int ln   = lane & 15;
  int sel  = lane >> 4;

  v8f acc[NS] = {};
  const _Float16* aRow = A + (size_t)(mBase + ln) * K + (sel << 3);
  const _Float16* bRow = Bt + (size_t)(nBase + ln) * K + (sel << 3);

  for (int k0 = 0; k0 < K; k0 += 32) {
    v8h alo = *(const v8h*)(aRow + k0);
    v8h ahi = *(const v8h*)(aRow + k0 + 16);
    v16h a;
#pragma unroll
    for (int i = 0; i < 8; ++i) { a[i] = alo[i]; a[i + 8] = ahi[i]; }
#pragma unroll
    for (int s4 = 0; s4 < NS; ++s4) {
      v8h blo = *(const v8h*)(bRow + (size_t)(s4 << 4) * K + k0);
      v8h bhi = *(const v8h*)(bRow + (size_t)(s4 << 4) * K + k0 + 16);
      v16h b;
#pragma unroll
      for (int i = 0; i < 8; ++i) { b[i] = blo[i]; b[i + 8] = bhi[i]; }
      acc[s4] = __builtin_amdgcn_wmma_f32_16x16x32_f16(
          false, a, false, b, (short)0, acc[s4], false, false);
    }
  }

#pragma unroll
  for (int s4 = 0; s4 < NS; ++s4) {
    int n = nBase + (s4 << 4) + ln;
    float bv = 0.0f;
    if (bias && n < Nreal) bv = bias[n];
#pragma unroll
    for (int r = 0; r < 8; ++r) {
      int rowm = mBase + (sel << 3) + r;
      float v = acc[s4][r] + bv;
      if (Hout) Hout[(size_t)rowm * Npad + n] = (_Float16)v;
      if (Cout && n < Nreal) Cout[(size_t)rowm * ldOut + colOff + n] = v;
    }
  }
}

// ---------------- reductions ----------------

__device__ __forceinline__ float wave_max(float v) {
#pragma unroll
  for (int o = 16; o > 0; o >>= 1) v = fmaxf(v, __shfl_xor(v, o, 32));
  return v;
}
__device__ __forceinline__ float wave_sum(float v) {
#pragma unroll
  for (int o = 16; o > 0; o >>= 1) v += __shfl_xor(v, o, 32);
  return v;
}
__device__ __forceinline__ float block_max(float v, float* s) {
  int lane = threadIdx.x & 31, w = threadIdx.x >> 5;
  v = wave_max(v);
  if (lane == 0) s[w] = v;
  __syncthreads();
  float r = (lane < 8) ? s[lane] : -INFINITY;
  r = wave_max(r);
  if (threadIdx.x == 0) s[0] = r;
  __syncthreads();
  float out = s[0];
  __syncthreads();
  return out;
}
__device__ __forceinline__ float block_sum(float v, float* s) {
  int lane = threadIdx.x & 31, w = threadIdx.x >> 5;
  v = wave_sum(v);
  if (lane == 0) s[w] = v;
  __syncthreads();
  float r = (lane < 8) ? s[lane] : 0.0f;
  r = wave_sum(r);
  if (threadIdx.x == 0) s[0] = r;
  __syncthreads();
  float out = s[0];
  __syncthreads();
  return out;
}

// Online (max, sumexp) accumulation for one new value.
__device__ __forceinline__ void online_acc(float v, float& m, float& s) {
  float nm = fmaxf(m, v);
  s = s * expf(m - nm) + expf(v - nm);
  m = nm;
}

// ---------------- root softmax ----------------
// One block per row: softmax over 2002 root logits; write first 2000 probs,
// store the 2 cluster log-probs.
__global__ __launch_bounds__(256) void root_softmax_kernel(
    const float* __restrict__ rl, float* __restrict__ out, float* __restrict__ clust)
{
  __shared__ float s[8];
  int row = blockIdx.x;
  const float* r = rl + (size_t)row * HEAD_NPAD;
  float lm = -INFINITY, ls = 0.0f;
  for (int c = threadIdx.x; c < HEAD_N; c += blockDim.x) online_acc(r[c], lm, ls);
  float m = block_max(lm, s);
  float sum = block_sum(ls * expf(lm - m), s);
  float inv = 1.0f / sum;
  float* o = out + (size_t)row * VOCAB;
  for (int c = threadIdx.x; c < CUT0; c += blockDim.x) o[c] = expf(r[c] - m) * inv;
  if (threadIdx.x < 2) clust[row * 2 + threadIdx.x] = r[CUT0 + threadIdx.x] - m - logf(sum);
}

// ---------------- tail finalize (in place over d_out region) ----------------
// out = softmax(logits) * exp(cluster_logprob); one online pass + one write pass.
__global__ __launch_bounds__(256) void tail_finalize_kernel(
    float* __restrict__ out, const float* __restrict__ clust,
    int colOff, int ncols, int cidx)
{
  __shared__ float s[8];
  int row = blockIdx.x;
  float* p = out + (size_t)row * VOCAB + colOff;
  float lm = -INFINITY, ls = 0.0f;
  for (int c = threadIdx.x; c < ncols; c += blockDim.x) online_acc(p[c], lm, ls);
  float m = block_max(lm, s);
  float sum = block_sum(ls * expf(lm - m), s);
  float scale = expf(clust[row * 2 + cidx]) / sum;
  for (int c = threadIdx.x; c < ncols; c += blockDim.x) p[c] = expf(p[c] - m) * scale;
}

// ---------------- launcher ----------------

static inline size_t align256(size_t x) { return (x + 255) & ~(size_t)255; }

extern "C" void kernel_launch(void* const* d_in, const int* in_sizes, int n_in,
                              void* d_out, int out_size, void* d_ws, size_t ws_size,
                              hipStream_t stream) {
  const float* x    = (const float*)d_in[0];
  // d_in[1] = targets (unused by the reference computation)
  const float* hk   = (const float*)d_in[2];
  const float* pk0  = (const float*)d_in[3];
  const float* pb0  = (const float*)d_in[4];
  const float* sk0  = (const float*)d_in[5];
  const float* sb0  = (const float*)d_in[6];
  const float* pk1  = (const float*)d_in[7];
  const float* pb1  = (const float*)d_in[8];
  const float* sk1  = (const float*)d_in[9];
  const float* sb1  = (const float*)d_in[10];
  float* out = (float*)d_out;

  char* ws = (char*)d_ws;
  size_t off = 0;
  auto carve = [&](size_t bytes) { char* p = ws + off; off += align256(bytes); return p; };

  _Float16* xh     = (_Float16*)carve((size_t)ROWS * DMODEL * 2);
  _Float16* headT  = (_Float16*)carve((size_t)HEAD_NPAD * DMODEL * 2);
  _Float16* pk0T   = (_Float16*)carve((size_t)256 * DMODEL * 2);
  _Float16* pk1T   = (_Float16*)carve((size_t)64 * DMODEL * 2);
  _Float16* sk0T   = (_Float16*)carve((size_t)TAIL0_NP * 256 * 2);
  _Float16* sk1T   = (_Float16*)carve((size_t)TAIL1_NP * 64 * 2);
  float*    rootL  = (float*)   carve((size_t)ROWS * HEAD_NPAD * 4);
  _Float16* h0h    = (_Float16*)carve((size_t)ROWS * 256 * 2);
  _Float16* h1h    = (_Float16*)carve((size_t)ROWS * 64 * 2);
  float*    clust  = (float*)   carve((size_t)ROWS * 2 * 4);
  (void)ws_size;

  // 1) precision conversion + weight transposes (f32 -> f16, B transposed/padded)
  {
    int n = ROWS * DMODEL;
    cvt_f32_to_f16<<<(n + 255) / 256, 256, 0, stream>>>(x, xh, n);
  }
  {
    int n = HEAD_NPAD * DMODEL;
    cvt_transpose_f16<<<(n + 255) / 256, 256, 0, stream>>>(hk, headT, DMODEL, HEAD_N, HEAD_NPAD);
  }
  {
    int n = 256 * DMODEL;
    cvt_transpose_f16<<<(n + 255) / 256, 256, 0, stream>>>(pk0, pk0T, DMODEL, 256, 256);
  }
  {
    int n = 64 * DMODEL;
    cvt_transpose_f16<<<(n + 255) / 256, 256, 0, stream>>>(pk1, pk1T, DMODEL, 64, 64);
  }
  {
    int n = TAIL0_NP * 256;
    cvt_transpose_f16<<<(n + 255) / 256, 256, 0, stream>>>(sk0, sk0T, 256, TAIL0_N, TAIL0_NP);
  }
  {
    int n = TAIL1_NP * 64;
    cvt_transpose_f16<<<(n + 255) / 256, 256, 0, stream>>>(sk1, sk1T, 64, TAIL1_N, TAIL1_NP);
  }

  auto gemm_blocks = [](int M, int Npad, int ns) {
    int waves = (M >> 4) * (Npad / (16 * ns));
    return (waves + 7) / 8;
  };

  // 2) root logits: [2048x1024]*[1024x2002] -> ws (f32)
  gemm_wmma_f16<8><<<gemm_blocks(ROWS, HEAD_NPAD, 8), 256, 0, stream>>>(
      xh, headT, nullptr, rootL, nullptr, ROWS, DMODEL, HEAD_NPAD, HEAD_N, HEAD_NPAD, 0);

  // 3) tail projections h_i = x @ pk_i + pb_i (stored f16 for next GEMM)
  gemm_wmma_f16<8><<<gemm_blocks(ROWS, 256, 8), 256, 0, stream>>>(
      xh, pk0T, pb0, nullptr, h0h, ROWS, DMODEL, 256, 256, 0, 0);
  gemm_wmma_f16<4><<<gemm_blocks(ROWS, 64, 4), 256, 0, stream>>>(
      xh, pk1T, pb1, nullptr, h1h, ROWS, DMODEL, 64, 64, 0, 0);

  // 4) tail logits straight into d_out at their final column offsets
  gemm_wmma_f16<8><<<gemm_blocks(ROWS, TAIL0_NP, 8), 256, 0, stream>>>(
      h0h, sk0T, sb0, out, nullptr, ROWS, 256, TAIL0_NP, TAIL0_N, VOCAB, CUT0);
  gemm_wmma_f16<8><<<gemm_blocks(ROWS, TAIL1_NP, 8), 256, 0, stream>>>(
      h1h, sk1T, sb1, out, nullptr, ROWS, 64, TAIL1_NP, TAIL1_N, VOCAB, 10000);

  // 5) root softmax -> out[:, :2000] + cluster log-probs
  root_softmax_kernel<<<ROWS, 256, 0, stream>>>(rootL, out, clust);

  // 6) tail softmax in place, scaled by cluster probability
  tail_finalize_kernel<<<ROWS, 256, 0, stream>>>(out, clust, CUT0, TAIL0_N, 0);
  tail_finalize_kernel<<<ROWS, 256, 0, stream>>>(out, clust, 10000, TAIL1_N, 1);
}
